// FlashAttention_45844480917693
// MI455X (gfx1250) — matrix operand order repaired and checked by reading, not compiled, not run on hardware
//
#include <hip/hip_runtime.h>
#include <hip/hip_bf16.h>
#include <cstdint>

// ---------------------------------------------------------------------------
// Cosine-sim attention (b=2, n=4096, DIM=1024, HEADS=8, DHEAD=64, SCALE=16)
// All matmuls via v_wmma_f32_16x16x32_f16; attention K/V tiles double-buffered
// through global_load_async_to_lds_b128 (ASYNCcnt); V transposed once in the
// projection epilogue so the hot loop has zero transpose work.
// ---------------------------------------------------------------------------

typedef __attribute__((ext_vector_type(16))) _Float16 v16h;
typedef __attribute__((ext_vector_type(8)))  _Float16 v8h;
typedef __attribute__((ext_vector_type(8)))  float    v8f;

#define B_       2
#define N_       4096
#define ROWS_    (B_ * N_)      // 8192
#define DIM_     1024
#define HEADS_   8
#define DHEAD_   64
#define INNER_   512
#define SCALE_   16.0f
#define EPS_     1e-6f

__device__ __forceinline__ v8f wmma16(v16h a, v16h b, v8f c) {
  return __builtin_amdgcn_wmma_f32_16x16x32_f16(
      false, a, false, b, (short)0, c, false, false);
}

// A fragment (16x32 f16, MxK): lane holds row m=lane%16; K chunks
// [base, base+8) and [base+16, base+24), base = (lane>>4)*8.
__device__ __forceinline__ v16h load_a_frag(const _Float16* lds, int stride, int lane) {
  const int row = lane & 15;
  const int kb  = (lane >> 4) * 8;
  const _Float16* p = lds + row * stride + kb;
  v16h a;
#pragma unroll
  for (int i = 0; i < 8; ++i) { a[i] = p[i]; a[i + 8] = p[i + 16]; }
  return a;
}

// B fragment (32x16 f16, KxN) from N-major LDS (Bt[n][k]): lane n=lane%16
// reads K chunks [base, base+8) and [base+16, base+24), base = (lane>>4)*8.
__device__ __forceinline__ v16h load_b_frag(const _Float16* ldsBt, int stride, int lane) {
  const int n  = lane & 15;
  const int kb = (lane >> 4) * 8;
  const _Float16* p = ldsBt + n * stride + kb;
  v16h b;
#pragma unroll
  for (int i = 0; i < 8; ++i) { b[i] = p[i]; b[i + 8] = p[i + 16]; }
  return b;
}

// Cross-lane XOR-add via ds_swizzle (group-of-32: offset = xor<<10 | and 0x1F).
template <int MASK>
__device__ __forceinline__ float xadd(float v) {
  const int s = __builtin_amdgcn_ds_swizzle(__float_as_int(v), (MASK << 10) | 0x1F);
  return v + __int_as_float(s);
}

// LDS byte offset of a shared-memory pointer (VDST operand of async loads).
__device__ __forceinline__ unsigned lds_off(const void* p) {
  return (unsigned)(uintptr_t)(__attribute__((address_space(3))) void*)(void*)p;
}

// Async global -> LDS, 16B per lane, tracked by ASYNCcnt.
__device__ __forceinline__ void async_b128(unsigned ldsoff, const void* gaddr) {
  asm volatile("global_load_async_to_lds_b128 %0, %1, off"
               :: "v"(ldsoff), "v"(gaddr) : "memory");
}
__device__ __forceinline__ void wait_async0() {
  asm volatile("s_wait_asynccnt 0x0" ::: "memory");
}

// ---------------------------------------------------------------------------
// Kernel 1: fused projections. grid (64, 24), block 256 (8 waves).
//  blockIdx.y 0..7  -> Q head tiles (l2-normalized, row-major out)
//  blockIdx.y 8..15 -> K head tiles (l2-normalized, row-major out)
//  blockIdx.y 16..23-> V head tiles (written TRANSPOSED: Vt[(b*8+h)][d][n])
// ---------------------------------------------------------------------------
__global__ __launch_bounds__(256) void proj_kernel(
    const float* __restrict__ X, const float* __restrict__ Wq,
    const float* __restrict__ Wkv,
    _Float16* __restrict__ Qo, _Float16* __restrict__ Ko, _Float16* __restrict__ Vt) {
  __shared__ _Float16 lds_x[128 * 40];   // 128 rows x 32 k (pad 40)
  __shared__ _Float16 lds_w[64 * 40];    // N-major: 64 n x 32 k (pad 40)
  __shared__ _Float16 lds_t[64 * 136];   // V transpose staging: 64 d x 128 j

  const int tid  = threadIdx.x;
  const int lane = tid & 31;
  const int wave = tid >> 5;
  const int m0   = blockIdx.x * 128;
  const int ct   = blockIdx.y;

  const float* Wsrc; int ldw; int ncol0; bool isV;
  if (ct < 8)       { Wsrc = Wq;  ldw = INNER_;   ncol0 = ct * 64;             isV = false; }
  else if (ct < 16) { Wsrc = Wkv; ldw = 2*INNER_; ncol0 = (ct - 8) * 64;       isV = false; }
  else              { Wsrc = Wkv; ldw = 2*INNER_; ncol0 = INNER_ + (ct-16)*64; isV = true;  }

  v8f acc[4] = {};

  for (int k0 = 0; k0 < DIM_; k0 += 32) {
    __syncthreads();
    const int xr  = tid >> 1;
    const int xc0 = (tid & 1) * 16;
    const int wk  = tid >> 3;
    const int wn0 = (tid & 7) * 8;
    { // X tile 128x32 f32 -> f16 LDS
      const float4* src = (const float4*)(X + (size_t)(m0 + xr) * DIM_ + k0 + xc0);
      _Float16* dst = &lds_x[xr * 40 + xc0];
#pragma unroll
      for (int v = 0; v < 4; ++v) {
        float4 f = src[v];
        dst[v*4+0] = (_Float16)f.x; dst[v*4+1] = (_Float16)f.y;
        dst[v*4+2] = (_Float16)f.z; dst[v*4+3] = (_Float16)f.w;
      }
    }
    { // W tile 32k x 64n -> N-major f16 LDS
      const float4* src = (const float4*)(Wsrc + (size_t)(k0 + wk) * ldw + ncol0 + wn0);
      const float4 f0 = src[0], f1 = src[1];
      const float fv[8] = {f0.x, f0.y, f0.z, f0.w, f1.x, f1.y, f1.z, f1.w};
#pragma unroll
      for (int i = 0; i < 8; ++i) lds_w[(wn0 + i) * 40 + wk] = (_Float16)fv[i];
    }
    __syncthreads();
    if (k0 + 32 < DIM_) {
      __builtin_prefetch(X + (size_t)(m0 + xr) * DIM_ + k0 + 32 + xc0, 0, 0);
      __builtin_prefetch(Wsrc + (size_t)(k0 + 32 + wk) * ldw + ncol0 + wn0, 0, 0);
    }

    const v16h a = load_a_frag(&lds_x[wave * 16 * 40], 40, lane);
#pragma unroll
    for (int t = 0; t < 4; ++t) {
      const v16h b = load_b_frag(&lds_w[t * 16 * 40], 40, lane);
      acc[t] = wmma16(a, b, acc[t]);
    }
  }

  const int rb  = (lane >> 4) * 8;
  const int col = lane & 15;

  if (!isV) {
    // l2-normalize rows (tile is exactly one head's 64 columns)
#pragma unroll
    for (int r = 0; r < 8; ++r) {
      float ss = 0.0f;
#pragma unroll
      for (int t = 0; t < 4; ++t) ss += acc[t][r] * acc[t][r];
      ss = xadd<1>(ss); ss = xadd<2>(ss); ss = xadd<4>(ss); ss = xadd<8>(ss);
      const float inv = 1.0f / fmaxf(sqrtf(ss), 1e-12f);
#pragma unroll
      for (int t = 0; t < 4; ++t) acc[t][r] *= inv;
    }
    _Float16* Out = (ct < 8) ? Qo : Ko;
    const int outcol0 = (ct & 7) * 64;
#pragma unroll
    for (int t = 0; t < 4; ++t)
#pragma unroll
      for (int r = 0; r < 8; ++r)
        Out[(size_t)(m0 + wave * 16 + rb + r) * INNER_ + outcol0 + t * 16 + col] =
            (_Float16)acc[t][r];
  } else {
    // stage transpose in LDS (consecutive r -> consecutive addresses), then
    // write Vt[(b*8+h)][d][n] fully coalesced.
    __syncthreads();
#pragma unroll
    for (int t = 0; t < 4; ++t)
#pragma unroll
      for (int r = 0; r < 8; ++r)
        lds_t[(t * 16 + col) * 136 + wave * 16 + rb + r] = (_Float16)acc[t][r];
    __syncthreads();

    const int bh = (m0 >> 12) * HEADS_ + (ct - 16);
    const int n0 = m0 & (N_ - 1);
    const int d  = tid >> 2;
    const int c0 = (tid & 3) * 32;
    const _Float16* src = &lds_t[d * 136 + c0];
    _Float16* dst = Vt + ((size_t)bh * DHEAD_ + d) * N_ + n0 + c0;
#pragma unroll
    for (int c = 0; c < 4; ++c)
      *(v8h*)(dst + c * 8) = *(const v8h*)(src + c * 8);
  }
}

// ---------------------------------------------------------------------------
// Kernel 2: attention per (b,h). grid (32, 16), block 256 (8 waves).
// logits = 16*(q.k)-16 <= 0 -> exp stable, no running max needed.
// K and Vt tiles double-buffered via async global->LDS loads.
// ---------------------------------------------------------------------------
__global__ __launch_bounds__(256) void attn_kernel(
    const _Float16* __restrict__ Q, const _Float16* __restrict__ K,
    const _Float16* __restrict__ Vt, _Float16* __restrict__ O) {
  __shared__ _Float16 lds_k[2][32 * 72];     // K rows (N-major for QK^T B)
  __shared__ _Float16 lds_vt[2][64 * 40];    // Vt rows (N-major for PV B)
  __shared__ _Float16 lds_p[8 * 16 * 40];    // per-wave P staging (C->A relayout)

  const int tid  = threadIdx.x;
  const int lane = tid & 31;
  const int wave = tid >> 5;
  const int i0   = blockIdx.x * 128;
  const int bh   = blockIdx.y;
  const size_t rowbase = (size_t)(bh >> 3) * N_;
  const int dcol0 = (bh & 7) * DHEAD_;

  // Resident Q A-fragments (d=64 -> two k-steps of 32), loaded as 2x b128.
  v16h aq[2];
  {
    const int row = i0 + wave * 16 + (lane & 15);
    const _Float16* qp = Q + (rowbase + row) * INNER_ + dcol0;
#pragma unroll
    for (int ks = 0; ks < 2; ++ks) {
      const int base = (lane >> 4) * 8 + ks * 32;
      const v8h q0 = *(const v8h*)(qp + base);
      const v8h q1 = *(const v8h*)(qp + base + 16);
#pragma unroll
      for (int i = 0; i < 8; ++i) { aq[ks][i] = q0[i]; aq[ks][i + 8] = q1[i]; }
    }
  }

  // Per-thread async-load slots: K tile 32x64 (16B/lane), Vt tile 64x32.
  const int kn = tid >> 3, kd = (tid & 7) * 8;
  const int vd = tid >> 2, vc = (tid & 3) * 8;
  const _Float16* gK = K + (rowbase + kn) * INNER_ + dcol0 + kd;
  const _Float16* gV = Vt + ((size_t)bh * DHEAD_ + vd) * N_ + vc;
  const unsigned ldsKoff[2] = { lds_off(&lds_k[0][kn * 72 + kd]),
                                lds_off(&lds_k[1][kn * 72 + kd]) };
  const unsigned ldsVoff[2] = { lds_off(&lds_vt[0][vd * 40 + vc]),
                                lds_off(&lds_vt[1][vd * 40 + vc]) };

  v8f acc[4] = {};
  float rs[8] = {};
  _Float16* myP = &lds_p[wave * 16 * 40];

  // Prologue: kick off tile 0.
  async_b128(ldsKoff[0], gK);
  async_b128(ldsVoff[0], gV);

  int buf = 0;
  for (int jt = 0; jt < N_ / 32; ++jt) {
    wait_async0();        // my async writes for tile jt have landed
    __syncthreads();      // everyone's landed; prev compute done -> buf^1 free

    if (jt + 1 < N_ / 32) {   // overlap next tile's loads with this compute
      const int j1 = (jt + 1) * 32;
      async_b128(ldsKoff[buf ^ 1], gK + (size_t)j1 * INNER_);
      async_b128(ldsVoff[buf ^ 1], gV + j1);
    }

    // S = Q K^T (16x32 per wave) -> P = exp(16S-16) -> rowsum + LDS stage
#pragma unroll
    for (int ns = 0; ns < 2; ++ns) {
      v8f s = {};
#pragma unroll
      for (int ks = 0; ks < 2; ++ks) {
        const v16h bk = load_b_frag(&lds_k[buf][ns * 16 * 72 + ks * 32], 72, lane);
        s = wmma16(aq[ks], bk, s);
      }
      const int col = ns * 16 + (lane & 15);
      const int rb  = (lane >> 4) * 8;
#pragma unroll
      for (int r = 0; r < 8; ++r) {
        const float pv = __expf(SCALE_ * s[r] - SCALE_);
        rs[r] += pv;
        myP[(rb + r) * 40 + col] = (_Float16)pv;
      }
    }

    // P (A-fragment, k = 32 keys) x V^T -> accumulate O
    const v16h ap = load_a_frag(myP, 40, lane);
#pragma unroll
    for (int t = 0; t < 4; ++t) {
      const v16h bv = load_b_frag(&lds_vt[buf][t * 16 * 40], 40, lane);
      acc[t] = wmma16(ap, bv, acc[t]);
    }
    buf ^= 1;
  }

  // finalize: divide by clip(rowsum, EPS)
#pragma unroll
  for (int r = 0; r < 8; ++r) {
    float s = rs[r];
    s = xadd<1>(s); s = xadd<2>(s); s = xadd<4>(s); s = xadd<8>(s);
    const float inv = 1.0f / fmaxf(s, EPS_);
#pragma unroll
    for (int t = 0; t < 4; ++t) acc[t][r] *= inv;
  }

  const int rb  = (lane >> 4) * 8;
  const int col = lane & 15;
#pragma unroll
  for (int t = 0; t < 4; ++t)
#pragma unroll
    for (int r = 0; r < 8; ++r)
      O[(rowbase + i0 + wave * 16 + rb + r) * INNER_ + dcol0 + t * 16 + col] =
          (_Float16)acc[t][r];
}

// ---------------------------------------------------------------------------
// Kernel 3: out = attn(8192x512 f16) @ Wo(512x1024). grid (64, 16), block 256.
// ---------------------------------------------------------------------------
__global__ __launch_bounds__(256) void outproj_kernel(
    const _Float16* __restrict__ A, const float* __restrict__ Wo,
    float* __restrict__ Out) {
  __shared__ _Float16 lds_a[128 * 40];
  __shared__ _Float16 lds_w[64 * 40];

  const int tid  = threadIdx.x;
  const int lane = tid & 31;
  const int wave = tid >> 5;
  const int m0    = blockIdx.x * 128;
  const int ncol0 = blockIdx.y * 64;

  v8f acc[4] = {};

  for (int k0 = 0; k0 < INNER_; k0 += 32) {
    __syncthreads();
    const int ar  = tid >> 1;
    const int ac0 = (tid & 1) * 16;
    const int wk  = tid >> 3;
    const int wn0 = (tid & 7) * 8;
    { // A tile 128x32 f16
      const v8h* src = (const v8h*)(A + (size_t)(m0 + ar) * INNER_ + k0 + ac0);
      *(v8h*)&lds_a[ar * 40 + ac0]     = src[0];
      *(v8h*)&lds_a[ar * 40 + ac0 + 8] = src[1];
    }
    { // Wo tile 32k x 64n -> N-major f16
      const float4* src = (const float4*)(Wo + (size_t)(k0 + wk) * DIM_ + ncol0 + wn0);
      const float4 f0 = src[0], f1 = src[1];
      const float fv[8] = {f0.x, f0.y, f0.z, f0.w, f1.x, f1.y, f1.z, f1.w};
#pragma unroll
      for (int i = 0; i < 8; ++i) lds_w[(wn0 + i) * 40 + wk] = (_Float16)fv[i];
    }
    __syncthreads();
    if (k0 + 32 < INNER_) {
      __builtin_prefetch(A + (size_t)(m0 + ar) * INNER_ + k0 + 32 + ac0, 0, 0);
      __builtin_prefetch(Wo + (size_t)(k0 + 32 + wk) * DIM_ + ncol0 + wn0, 0, 0);
    }

    const v16h a = load_a_frag(&lds_a[wave * 16 * 40], 40, lane);
#pragma unroll
    for (int t = 0; t < 4; ++t) {
      const v16h b = load_b_frag(&lds_w[t * 16 * 40], 40, lane);
      acc[t] = wmma16(a, b, acc[t]);
    }
  }

  const int rb  = (lane >> 4) * 8;
  const int col = lane & 15;
#pragma unroll
  for (int t = 0; t < 4; ++t)
#pragma unroll
    for (int r = 0; r < 8; ++r)
      Out[(size_t)(m0 + wave * 16 + rb + r) * DIM_ + ncol0 + t * 16 + col] = acc[t][r];
}

// ---------------------------------------------------------------------------
extern "C" void kernel_launch(void* const* d_in, const int* in_sizes, int n_in,
                              void* d_out, int out_size, void* d_ws, size_t ws_size,
                              hipStream_t stream) {
  const float* X   = (const float*)d_in[0];
  const float* Wq  = (const float*)d_in[1];
  const float* Wkv = (const float*)d_in[2];
  const float* Wo  = (const float*)d_in[3];
  float* out       = (float*)d_out;

  // Workspace: Q | K | Vt | attn, each ROWS_*INNER_ f16 (8 MB) = 32 MB total.
  _Float16* ws = (_Float16*)d_ws;
  _Float16* Qb  = ws;
  _Float16* Kb  = Qb + (size_t)ROWS_ * INNER_;
  _Float16* Vtb = Kb + (size_t)ROWS_ * INNER_;
  _Float16* Ab  = Vtb + (size_t)ROWS_ * INNER_;

  proj_kernel<<<dim3(ROWS_ / 128, 24), 256, 0, stream>>>(X, Wq, Wkv, Qb, Kb, Vtb);
  attn_kernel<<<dim3(N_ / 128, B_ * HEADS_), 256, 0, stream>>>(Qb, Kb, Vtb, Ab);
  outproj_kernel<<<dim3(ROWS_ / 128, DIM_ / 64), 256, 0, stream>>>(Ab, Wo, out);
}
